// My_scaled_dot_product_attention_43662637531564
// MI455X (gfx1250) — matrix operand order repaired and checked by reading, not compiled, not run on hardware
//
#include <hip/hip_runtime.h>
#include <hip/hip_bf16.h>

// ---------------------------------------------------------------------------
// Flash attention (causal) for B=2, H=16, S=2048, D=64, fp32 I/O.
// f16 WMMA (v_wmma_f32_16x16x32_f16) for QK^T and PV, f32 accumulation,
// online softmax in the WMMA C-register layout, DPP16 row reductions,
// diagonal-split causal masking, double-buffered (ping-pong) K/V staging
// so tile t+1 global loads are in flight during tile t compute.
// Wave32 / gfx1250.
// ---------------------------------------------------------------------------

typedef __attribute__((ext_vector_type(16))) _Float16 v16h;
typedef __attribute__((ext_vector_type(8)))  float    v8f;
typedef __attribute__((ext_vector_type(4)))  float    v4f;

namespace {
constexpr int kS   = 2048;
constexpr int kD   = 64;
constexpr int kBM  = 128;       // query rows per workgroup (8 waves x 16 rows)
constexpr int kBKV = 64;        // keys per inner step
constexpr int kNW  = 8;         // waves per block
constexpr int kLDK = kD + 8;    // s_k row stride (halfs), 144B, 16B aligned
constexpr int kLDV = kBKV + 8;  // s_vt row stride (halfs), 144B
constexpr int kLDP = kBKV + 8;  // s_p row stride (halfs), 144B
constexpr int kKBuf = kBKV * kLDK;   // halfs per K buffer
constexpr int kVBuf = kBKV * kLDV;   // halfs per V buffer
}

// DPP16 lane permute within rows of 16 lanes (VALU, no LDS traffic).
template <int CTRL>
__device__ __forceinline__ float dpp_mov(float x) {
    return __builtin_bit_cast(
        float, __builtin_amdgcn_update_dpp(0, __builtin_bit_cast(int, x),
                                           CTRL, 0xf, 0xf, true));
}
// Butterfly reduce-to-all across the 16 lanes of a DPP row.
__device__ __forceinline__ float row16_max(float x) {
    x = fmaxf(x, dpp_mov<0xB1>(x));    // quad_perm(1,0,3,2)
    x = fmaxf(x, dpp_mov<0x4E>(x));    // quad_perm(2,3,0,1)
    x = fmaxf(x, dpp_mov<0x141>(x));   // row_half_mirror
    x = fmaxf(x, dpp_mov<0x140>(x));   // row_mirror
    return x;
}
__device__ __forceinline__ float row16_sum(float x) {
    x += dpp_mov<0xB1>(x);
    x += dpp_mov<0x4E>(x);
    x += dpp_mov<0x141>(x);
    x += dpp_mov<0x140>(x);
    return x;
}

// One 64-key flash-attention step for one wave's 16 query rows.
// MASKED=false: tile is provably fully below the causal diagonal.
template <bool MASKED>
__device__ __forceinline__ void fa_step(const _Float16* __restrict__ s_k,
                                        const _Float16* __restrict__ s_vt,
                                        _Float16* __restrict__ pw,
                                        const v16h& a0, const v16h& a1,
                                        v8f (&acc)[4],
                                        float (&mrow)[8], float (&lrow)[8],
                                        int kv0, int q0, int half, int l15)
{
    // ---- Scores: four 16x16 tiles, each = 2 WMMAs over D=64 ----
    v8f sc[4];
#pragma unroll
    for (int j = 0; j < 4; ++j) {
        // B operand in this code's k order (same as A): lane -> col n = l15; per-lane contiguous
        // K(depth) range starting at half*16 (+32 for the second WMMA).
        const _Float16* kr = &s_k[(j * 16 + l15) * kLDK + half * 16];
        v16h b0, b1;
#pragma unroll
        for (int v = 0; v < 8; ++v) {
            b0[2 * v]     = kr[2 * v];
            b0[2 * v + 1] = kr[2 * v + 1];
            b1[2 * v]     = kr[32 + 2 * v];
            b1[2 * v + 1] = kr[32 + 2 * v + 1];
        }
        v8f c = {};
        c = __builtin_amdgcn_wmma_f32_16x16x32_f16(false, a0, false, b0,
                                                   (short)0, c, false, false);
        c = __builtin_amdgcn_wmma_f32_16x16x32_f16(false, a1, false, b1,
                                                   (short)0, c, false, false);
        sc[j] = c;
    }

    // ---- Online softmax in C layout + stage P (f16) to LDS ----
#pragma unroll
    for (int r = 0; r < 8; ++r) {
        float x[4];
#pragma unroll
        for (int j = 0; j < 4; ++j) x[j] = sc[j][r];
        if (MASKED) {
            const int qm = q0 + r + 8 * half;      // query row index
#pragma unroll
            for (int j = 0; j < 4; ++j) {
                const int kn = kv0 + j * 16 + l15; // key column index
                if (kn > qm) x[j] = -__builtin_inff();
            }
        }

        const float mx = row16_max(fmaxf(fmaxf(x[0], x[1]), fmaxf(x[2], x[3])));
        const float mi  = fmaxf(mrow[r], mx);
        const float rsc = __expf(mrow[r] - mi);
        float p[4];
#pragma unroll
        for (int j = 0; j < 4; ++j) p[j] = __expf(x[j] - mi);
        const float rs = row16_sum((p[0] + p[1]) + (p[2] + p[3]));

        lrow[r] = lrow[r] * rsc + rs;
        mrow[r] = mi;
#pragma unroll
        for (int nb = 0; nb < 4; ++nb) acc[nb][r] *= rsc;

        const int prow = r + 8 * half;
#pragma unroll
        for (int j = 0; j < 4; ++j)
            pw[prow * kLDP + j * 16 + l15] = (_Float16)p[j];
    }

    // Same-wave LDS write -> cross-lane-read ordering (DS ops are in-order
    // per wave; wait drains the stores before dependent loads issue).
    asm volatile("s_wait_dscnt 0" ::: "memory");

    // ---- Reload P in A-operand layout (two K=32 halves) ----
    v16h pa0, pa1;
    {
        const _Float16* pr = &pw[l15 * kLDP];
#pragma unroll
        for (int v = 0; v < 8; ++v) {
            const int c2 = half * 16 + 2 * v;
            pa0[2 * v]     = pr[c2];
            pa0[2 * v + 1] = pr[c2 + 1];
            pa1[2 * v]     = pr[32 + c2];
            pa1[2 * v + 1] = pr[32 + c2 + 1];
        }
    }

    // ---- PV: 4 n-blocks of 16 output columns, K=64 split as 32+32 ----
#pragma unroll
    for (int nb = 0; nb < 4; ++nb) {
        const _Float16* vr = &s_vt[(nb * 16 + l15) * kLDV + half * 16];
        v16h vb0, vb1;
#pragma unroll
        for (int v = 0; v < 8; ++v) {
            vb0[2 * v]     = vr[2 * v];
            vb0[2 * v + 1] = vr[2 * v + 1];
            vb1[2 * v]     = vr[32 + 2 * v];
            vb1[2 * v + 1] = vr[32 + 2 * v + 1];
        }
        acc[nb] = __builtin_amdgcn_wmma_f32_16x16x32_f16(false, pa0, false, vb0,
                                                         (short)0, acc[nb], false, false);
        acc[nb] = __builtin_amdgcn_wmma_f32_16x16x32_f16(false, pa1, false, vb1,
                                                         (short)0, acc[nb], false, false);
    }
}

__global__ __launch_bounds__(256)
void fa_causal_wmma_kernel(const float* __restrict__ Q,
                           const float* __restrict__ K,
                           const float* __restrict__ V,
                           float* __restrict__ O)
{
    __shared__ __align__(16) _Float16 s_k [2 * kKBuf];        // ping-pong K tiles
    __shared__ __align__(16) _Float16 s_vt[2 * kVBuf];        // ping-pong V^T tiles
    __shared__ __align__(16) _Float16 s_p [kNW * 16 * kLDP];  // per-wave P tile

    const int bh   = blockIdx.y;        // batch*head
    const int mt   = blockIdx.x;        // query tile index
    const int tid  = threadIdx.x;
    // Scalar wave id: the per-wave tile classification must be s_cbranch so
    // EXEC stays all-ones around the WMMAs (ISA requirement).
    const int wave = __builtin_amdgcn_readfirstlane(tid >> 5);
    const int lane = tid & 31;
    const int half = lane >> 4;
    const int l15  = lane & 15;

    const int q0 = mt * kBM + wave * 16;            // first query row of this wave
    const size_t base = (size_t)bh * kS * kD;
    const float* q_ptr = Q + base;
    const float* k_ptr = K + base;
    const float* v_ptr = V + base;
    float*       o_ptr = O + base;

    // Staging decomposition: 256 threads x 16 floats = 64x64 tile.
    const int stg_row = tid >> 2;          // 0..63
    const int stg_col = (tid & 3) * 16;    // 0,16,32,48

    // ---- Load Q stripe once, pre-scaled by 1/sqrt(D), into A-layout f16. ----
    v16h a0, a1;
    {
        const float* qr = q_ptr + (size_t)(q0 + l15) * kD;
        const float s = 0.125f;   // 1/sqrt(64)
#pragma unroll
        for (int v = 0; v < 8; ++v) {
            const int d = half * 16 + 2 * v;
            a0[2 * v]     = (_Float16)(qr[d] * s);
            a0[2 * v + 1] = (_Float16)(qr[d + 1] * s);
            a1[2 * v]     = (_Float16)(qr[32 + d] * s);
            a1[2 * v + 1] = (_Float16)(qr[32 + d + 1] * s);
        }
    }

    v8f acc[4] = {};
    float mrow[8], lrow[8];
#pragma unroll
    for (int r = 0; r < 8; ++r) { mrow[r] = -__builtin_inff(); lrow[r] = 0.0f; }

    const int nT      = 2 * (mt + 1);               // KV tiles for this block
    const int kv_diag = ((q0 + 15) / kBKV) * kBKV;  // the one masked tile (scalar)
    _Float16* pw = &s_p[wave * 16 * kLDP];

    // ---- Prologue: load + store tile 0 into buffer 0 ----
    v4f kregs[4], vregs[4];
    {
        const v4f* k4 = (const v4f*)(k_ptr + (size_t)stg_row * kD + stg_col);
        const v4f* v4p = (const v4f*)(v_ptr + (size_t)stg_row * kD + stg_col);
#pragma unroll
        for (int c = 0; c < 4; ++c) { kregs[c] = k4[c]; vregs[c] = v4p[c]; }
#pragma unroll
        for (int c = 0; c < 4; ++c)
#pragma unroll
            for (int i = 0; i < 4; ++i) {
                s_k [stg_row * kLDK + stg_col + c * 4 + i]      = (_Float16)kregs[c][i];
                s_vt[(stg_col + c * 4 + i) * kLDV + stg_row]    = (_Float16)vregs[c][i];
            }
    }
    __syncthreads();

    for (int t = 0; t < nT; ++t) {
        const int buf  = t & 1;
        const int kv0  = t * kBKV;
        const bool has_next = (t + 1 < nT);

        // Issue tile t+1 global loads now; they complete during compute below.
        if (has_next) {
            const size_t off = (size_t)(kv0 + kBKV + stg_row) * kD + stg_col;
            const v4f* k4  = (const v4f*)(k_ptr + off);
            const v4f* v4p = (const v4f*)(v_ptr + off);
#pragma unroll
            for (int c = 0; c < 4; ++c) { kregs[c] = k4[c]; vregs[c] = v4p[c]; }
            if (t + 2 < nT) {               // prime L2 for tile t+2
                __builtin_prefetch(k_ptr + off + kBKV * kD, 0, 1);
                __builtin_prefetch(v_ptr + off + kBKV * kD, 0, 1);
            }
        }

        // Scalar three-way classification (uniform per wave).
        const _Float16* skb = &s_k [buf * kKBuf];
        const _Float16* svb = &s_vt[buf * kVBuf];
        if (kv0 < kv_diag) {
            fa_step<false>(skb, svb, pw, a0, a1, acc, mrow, lrow,
                           kv0, q0, half, l15);
        } else if (kv0 == kv_diag) {
            fa_step<true>(skb, svb, pw, a0, a1, acc, mrow, lrow,
                          kv0, q0, half, l15);
        }
        // else: tile fully above this wave's diagonal -> barrier only.

        // Convert + store tile t+1 into the other buffer. Safe: every wave
        // finished reading it (tile t-1) before the barrier that ended
        // iteration t-1.
        if (has_next) {
            _Float16* skn = &s_k [(buf ^ 1) * kKBuf];
            _Float16* svn = &s_vt[(buf ^ 1) * kVBuf];
#pragma unroll
            for (int c = 0; c < 4; ++c)
#pragma unroll
                for (int i = 0; i < 4; ++i) {
                    skn[stg_row * kLDK + stg_col + c * 4 + i]   = (_Float16)kregs[c][i];
                    svn[(stg_col + c * 4 + i) * kLDV + stg_row] = (_Float16)vregs[c][i];
                }
        }
        __syncthreads();
    }

    // ---- Epilogue: normalize by row sum, write f32 output ----
#pragma unroll
    for (int r = 0; r < 8; ++r) {
        const int row = q0 + r + 8 * half;
        const float inv = 1.0f / lrow[r];   // every row has >= 1 valid key
        float* orow = o_ptr + (size_t)row * kD;
#pragma unroll
        for (int nb = 0; nb < 4; ++nb)
            orow[nb * 16 + l15] = acc[nb][r] * inv;
    }
}

extern "C" void kernel_launch(void* const* d_in, const int* in_sizes, int n_in,
                              void* d_out, int out_size, void* d_ws, size_t ws_size,
                              hipStream_t stream) {
    (void)in_sizes; (void)n_in; (void)d_ws; (void)ws_size; (void)out_size;
    const float* q = (const float*)d_in[0];
    const float* k = (const float*)d_in[1];
    const float* v = (const float*)d_in[2];
    // d_in[3] is the causal mask; causality is applied analytically.
    float* out = (float*)d_out;

    const int BH = 2 * 16;                 // B*H
    dim3 grid(kS / kBM, BH);               // (16, 32)
    dim3 block(256);                       // 8 wave32 waves
    fa_causal_wmma_kernel<<<grid, block, 0, stream>>>(q, k, v, out);
}
